// WhisperDecoderLayer_7052336300498
// MI455X (gfx1250) — matrix operand order repaired and checked by reading, not compiled, not run on hardware
//
#include <hip/hip_runtime.h>
#include <hip/hip_bf16.h>
#include <math.h>

// Problem constants (from reference)
#define B_   8
#define Q_   256
#define C_   192
#define H_   20
#define D_   64
#define E_   1500
#define I_   5120
#define HID  1280
#define BQ   2048
#define SCALE_ 0.125f
#define LSELF 448            // Q_ + C_

typedef __attribute__((ext_vector_type(16))) __bf16 v16bf;
typedef __attribute__((ext_vector_type(8)))  float  v8f;
typedef __attribute__((ext_vector_type(8)))  unsigned short ushort8;
typedef unsigned short u16;
typedef unsigned long long u64;

union Frag16 {
    v16bf v;
    ushort8 h[2];
    u16 u[16];
};

__device__ __forceinline__ u16 f2bfbits(float f) {
    unsigned int x = __float_as_uint(f);
    unsigned int r = x + 0x7FFFu + ((x >> 16) & 1u);   // RNE
    return (u16)(r >> 16);
}
__device__ __forceinline__ float gelu_f(float v) {
    return 0.5f * v * (1.0f + erff(v * 0.70710678118654752f));
}
// low 32 bits of a generic pointer to LDS == byte offset within LDS allocation
__device__ __forceinline__ unsigned lds_off(const void* p) {
    return (unsigned)(size_t)p;
}
// gfx1250 async global->LDS (ASYNCcnt); see CDNA5 ISA
#define ASYNC_LDS_B128(loff, gaddr)                                        \
    asm volatile("global_load_async_to_lds_b128 %0, %1, off"               \
                 :: "v"(loff), "v"(gaddr) : "memory")
#define WAIT_ASYNC() asm volatile("s_wait_asynccnt 0x0" ::: "memory")

// GEMM epilogue mode bits
#define OUT_F32  1
#define OUT_BF16 2
#define DO_GELU  4

// ---------------------------------------------------------------------------
// fp32 -> bf16 elementwise (n must be multiple of 8)
// ---------------------------------------------------------------------------
__launch_bounds__(256)
__global__ void cvt_bf16_k(const float* __restrict__ s, u16* __restrict__ d, int n8)
{
    int i = blockIdx.x * 256 + threadIdx.x;
    if (i >= n8) return;
    const float4* sp = (const float4*)s + (size_t)i * 2;
    float4 a = sp[0], b = sp[1];
    ushort8 o;
    o[0] = f2bfbits(a.x); o[1] = f2bfbits(a.y); o[2] = f2bfbits(a.z); o[3] = f2bfbits(a.w);
    o[4] = f2bfbits(b.x); o[5] = f2bfbits(b.y); o[6] = f2bfbits(b.z); o[7] = f2bfbits(b.w);
    *((ushort8*)d + i) = o;
}

// ---------------------------------------------------------------------------
// LayerNorm over last dim, fp32 in -> bf16 out, one row per 256-thread block
// ---------------------------------------------------------------------------
__launch_bounds__(256)
__global__ void layernorm_k(const float* __restrict__ x,
                            const float* __restrict__ w,
                            const float* __restrict__ b,
                            u16* __restrict__ y, int ncols)
{
    int row = blockIdx.x;
    const float* xr = x + (size_t)row * ncols;
    u16* yr = y + (size_t)row * ncols;
    __shared__ float red0[8], red1[8];

    float s = 0.f, ss = 0.f;
    for (int c = threadIdx.x; c < ncols; c += 256) {
        float v = xr[c]; s += v; ss += v * v;
    }
    #pragma unroll
    for (int m = 16; m; m >>= 1) {
        s += __shfl_xor(s, m, 32); ss += __shfl_xor(ss, m, 32);
    }
    int wave = threadIdx.x >> 5, lane = threadIdx.x & 31;
    if (lane == 0) { red0[wave] = s; red1[wave] = ss; }
    __syncthreads();
    if (wave == 0) {
        s = red0[lane & 7]; ss = red1[lane & 7];
        #pragma unroll
        for (int m = 4; m; m >>= 1) {
            s += __shfl_xor(s, m, 32); ss += __shfl_xor(ss, m, 32);
        }
        if (lane == 0) { red0[0] = s; red1[0] = ss; }
    }
    __syncthreads();
    s = red0[0]; ss = red1[0];
    float inv_n = 1.0f / (float)ncols;
    float mu  = s * inv_n;
    float var = ss * inv_n - mu * mu;
    float inv = rsqrtf(var + 1e-5f);
    for (int c = threadIdx.x; c < ncols; c += 256)
        yr[c] = f2bfbits((xr[c] - mu) * inv * w[c] + b[c]);
}

// ---------------------------------------------------------------------------
// GEMM: out = (A_bf16[M,K] * W_bf16[N,K]^T + bias) * scale (+resid) [gelu]
// Block tile 64x128, 8 waves of 32x32, K-step 64.
// Double-buffered async LDS staging: next tile's transfers overlap WMMA.
// Requires M%64==0, N%128==0, K%64==0.
// ---------------------------------------------------------------------------
__launch_bounds__(256)
__global__ void gemm_bf16_k(const u16* __restrict__ A,
                            const u16* __restrict__ W,
                            const float* __restrict__ bias,
                            const float* __restrict__ resid,
                            float* __restrict__ outF,
                            u16* __restrict__ outB,
                            int M, int N, int K, float scale, int mode)
{
    __shared__ u16 As[2][64 * 64];    // 2 x 8 KB
    __shared__ u16 Bs[2][128 * 64];   // 2 x 16 KB

    int t    = threadIdx.x;
    int wave = t >> 5;
    int lane = t & 31;
    int half = lane >> 4;
    int lr   = lane & 15;

    int bm = blockIdx.x * 64;
    int bn = blockIdx.y * 128;
    int waveM = (wave >> 2) * 32;
    int waveN = (wave & 3) * 32;

    unsigned asOff[2] = { lds_off(As[0]), lds_off(As[1]) };
    unsigned bsOff[2] = { lds_off(Bs[0]), lds_off(Bs[1]) };

    auto stage = [&](int buf, int k0) {
        #pragma unroll
        for (int i = 0; i < 2; ++i) {
            int idx = i * 256 + t;
            int row = idx >> 3, col = (idx & 7) * 8;
            u64 ga = (u64)(size_t)(A + (size_t)(bm + row) * K + k0 + col);
            ASYNC_LDS_B128(asOff[buf] + idx * 16, ga);
        }
        #pragma unroll
        for (int i = 0; i < 4; ++i) {
            int idx = i * 256 + t;
            int row = idx >> 3, col = (idx & 7) * 8;
            u64 ga = (u64)(size_t)(W + (size_t)(bn + row) * K + k0 + col);
            ASYNC_LDS_B128(bsOff[buf] + idx * 16, ga);
        }
    };

    v8f acc[2][2] = {};

    int nk = K / 64;
    stage(0, 0);
    WAIT_ASYNC();
    __syncthreads();

    for (int ki = 0; ki < nk; ++ki) {
        int buf = ki & 1;
        if (ki + 1 < nk) stage(buf ^ 1, (ki + 1) * 64);   // overlap with WMMA

        #pragma unroll
        for (int kk = 0; kk < 2; ++kk) {
            Frag16 afr[2], bfr[2];
            #pragma unroll
            for (int i = 0; i < 2; ++i) {
                const u16* arow = &As[buf][(waveM + i * 16 + lr) * 64 + kk * 32];
                afr[i].h[0] = *(const ushort8*)(arow + half * 8);
                afr[i].h[1] = *(const ushort8*)(arow + 16 + half * 8);
            }
            #pragma unroll
            for (int j = 0; j < 2; ++j) {
                const u16* brow = &Bs[buf][(waveN + j * 16 + lr) * 64 + kk * 32];
                bfr[j].h[0] = *(const ushort8*)(brow + half * 8);
                bfr[j].h[1] = *(const ushort8*)(brow + 16 + half * 8);
            }
            #pragma unroll
            for (int i = 0; i < 2; ++i)
                #pragma unroll
                for (int j = 0; j < 2; ++j)
                    acc[i][j] = __builtin_amdgcn_wmma_f32_16x16x32_bf16(
                        false, afr[i].v, false, bfr[j].v,
                        (short)0, acc[i][j], false, false);
        }
        WAIT_ASYNC();        // next buffer's transfers complete
        __syncthreads();     // all waves done reading current buffer
    }

    // ---- epilogue ----
    #pragma unroll
    for (int i = 0; i < 2; ++i) {
        #pragma unroll
        for (int j = 0; j < 2; ++j) {
            int col = bn + waveN + j * 16 + lr;
            float bv = bias ? bias[col] : 0.f;
            #pragma unroll
            for (int r = 0; r < 8; ++r) {
                int row = bm + waveM + i * 16 + r + 8 * half;
                float v = (acc[i][j][r] + bv) * scale;
                if (resid) v += resid[(size_t)row * N + col];
                if (mode & DO_GELU) v = gelu_f(v);
                if (mode & OUT_F32)  outF[(size_t)row * N + col] = v;
                if (mode & OUT_BF16) outB[(size_t)row * N + col] = f2bfbits(v);
            }
        }
    }
}

// ---------------------------------------------------------------------------
// Build self-attn bf16 KV cache (B,H,448,64) from past (B,H,192,64) fp32
// and new (B*Q,HID) fp32. 8 d-elements per thread.
// ---------------------------------------------------------------------------
__launch_bounds__(256)
__global__ void build_cache_self_k(const float* __restrict__ pk,
                                   const float* __restrict__ pv,
                                   const float* __restrict__ kn,
                                   const float* __restrict__ vn,
                                   u16* __restrict__ kc, u16* __restrict__ vc)
{
    int i = blockIdx.x * 256 + threadIdx.x;
    int total = B_ * H_ * LSELF * 8;
    if (i >= total) return;
    int d8 = (i & 7) * 8;
    int j  = (i >> 3) % LSELF;
    int h  = ((i >> 3) / LSELF) % H_;
    int b  = (i >> 3) / (LSELF * H_);
    const float *ks, *vs;
    if (j < C_) {
        size_t o = (((size_t)(b * H_ + h)) * C_ + j) * 64 + d8;
        ks = pk + o; vs = pv + o;
    } else {
        size_t o = ((size_t)(b * Q_ + (j - C_))) * HID + h * 64 + d8;
        ks = kn + o; vs = vn + o;
    }
    size_t doff = (((size_t)(b * H_ + h)) * LSELF + j) * 64 + d8;
    ushort8 ko, vo;
    #pragma unroll
    for (int e = 0; e < 8; ++e) { ko[e] = f2bfbits(ks[e]); vo[e] = f2bfbits(vs[e]); }
    *(ushort8*)(kc + doff) = ko;
    *(ushort8*)(vc + doff) = vo;
}

// ---------------------------------------------------------------------------
// Flash attention, WMMA, bf16 KV cache (B,H,L,64). q fp32 pre-scaled.
// One wave per 16-query tile; block stages 32-key K/V chunks in LDS shared
// by all 8 waves (V transposed so PV B-fragments are contiguous).
// Double-buffered K/V staging (async K loads overlap softmax + WMMA).
// grid = (2, H, B), block = 256. Writes ctx as bf16 (B*Q, HID).
// ---------------------------------------------------------------------------
__launch_bounds__(256)
__global__ void attn_k(const float* __restrict__ q,
                       const u16* __restrict__ kc,
                       const u16* __restrict__ vc,
                       const float* __restrict__ mask,
                       u16* __restrict__ ctx,
                       int L)
{
    __shared__ u16 Ks[2][32 * 64];   // K chunk, row-major (key, d)   2 x 4 KB
    __shared__ u16 Vt[2][64 * 32];   // V chunk transposed (d, key)   2 x 4 KB
    __shared__ u16 Plds[8 * 512];    // per-wave P tiles 16x32        8 KB

    int t    = threadIdx.x;
    int wave = t >> 5;
    int lane = t & 31;
    int half = lane >> 4;
    int lr   = lane & 15;

    int qt = blockIdx.x * 8 + wave;
    int h  = blockIdx.y;
    int b  = blockIdx.z;
    int qrowbase = qt * 16;
    size_t bhL = (size_t)(b * H_ + h) * L;

    unsigned ksOff[2] = { lds_off(Ks[0]), lds_off(Ks[1]) };
    int srow = t >> 3, scol = (t & 7) * 8;   // staging assignment

    auto stageKV = [&](int buf, int kb) {
        if (kb + srow < L) {
            u64 ga = (u64)(size_t)(kc + (bhL + kb + srow) * 64 + scol);
            ASYNC_LDS_B128(ksOff[buf] + t * 16, ga);
        }
        ushort8 vv = {0, 0, 0, 0, 0, 0, 0, 0};
        if (kb + srow < L)
            vv = *(const ushort8*)(vc + (bhL + kb + srow) * 64 + scol);
        #pragma unroll
        for (int e = 0; e < 8; ++e)
            Vt[buf][(scol + e) * 32 + srow] = vv[e];
    };

    // Q fragments (A layout), two 32-wide d-chunks
    Frag16 qa[2];
    {
        const float* qrow = q + ((size_t)(b * Q_ + qrowbase + lr)) * HID + h * 64;
        #pragma unroll
        for (int c = 0; c < 2; ++c) {
            #pragma unroll
            for (int e = 0; e < 8; ++e)
                qa[c].u[e] = f2bfbits(qrow[c * 32 + half * 8 + e]);
            #pragma unroll
            for (int e = 0; e < 8; ++e)
                qa[c].u[8 + e] = f2bfbits(qrow[c * 32 + 16 + half * 8 + e]);
        }
    }

    float m_run[8], l_run[8];
    #pragma unroll
    for (int r = 0; r < 8; ++r) { m_run[r] = -3.0e38f; l_run[r] = 0.f; }
    v8f oacc[4] = {};

    int nkc = (L + 31) / 32;
    stageKV(0, 0);
    WAIT_ASYNC();
    __syncthreads();

    for (int kcix = 0; kcix < nkc; ++kcix) {
        int buf = kcix & 1;
        int kb = kcix * 32;
        if (kcix + 1 < nkc) stageKV(buf ^ 1, kb + 32);   // overlap

        // ---- S = Q K^T (two 16x16 tiles over 32 keys) ----
        v8f sacc[2] = {};
        #pragma unroll
        for (int st = 0; st < 2; ++st) {
            const u16* krow = &Ks[buf][(st * 16 + lr) * 64];
            #pragma unroll
            for (int c = 0; c < 2; ++c) {
                Frag16 bk;
                bk.h[0] = *(const ushort8*)(krow + c * 32 + half * 8);
                bk.h[1] = *(const ushort8*)(krow + c * 32 + 16 + half * 8);
                sacc[st] = __builtin_amdgcn_wmma_f32_16x16x32_bf16(
                    false, qa[c].v, false, bk.v, (short)0, sacc[st], false, false);
            }
        }

        // ---- mask / OOB + online softmax ----
        float pmax[8];
        #pragma unroll
        for (int r = 0; r < 8; ++r) {
            int qrow = qrowbase + r + 8 * half;
            #pragma unroll
            for (int st = 0; st < 2; ++st) {
                int key = kb + st * 16 + lr;
                float s = sacc[st][r];
                if (key >= L) s = -3.0e38f;
                else if (mask) s += mask[(((size_t)b * Q_) + qrow) * L + key];
                sacc[st][r] = s;
            }
            pmax[r] = fmaxf(sacc[0][r], sacc[1][r]);
        }
        #pragma unroll
        for (int mk = 1; mk < 16; mk <<= 1)
            #pragma unroll
            for (int r = 0; r < 8; ++r)
                pmax[r] = fmaxf(pmax[r], __shfl_xor(pmax[r], mk, 32));

        float psum[8];
        #pragma unroll
        for (int r = 0; r < 8; ++r) {
            float mnew = fmaxf(m_run[r], pmax[r]);
            float resc = __expf(m_run[r] - mnew);
            m_run[r] = mnew;
            l_run[r] *= resc;
            #pragma unroll
            for (int g = 0; g < 4; ++g) oacc[g][r] *= resc;
            float p0 = __expf(sacc[0][r] - mnew);
            float p1 = __expf(sacc[1][r] - mnew);
            sacc[0][r] = p0; sacc[1][r] = p1;
            psum[r] = p0 + p1;
        }
        #pragma unroll
        for (int mk = 1; mk < 16; mk <<= 1)
            #pragma unroll
            for (int r = 0; r < 8; ++r)
                psum[r] += __shfl_xor(psum[r], mk, 32);
        #pragma unroll
        for (int r = 0; r < 8; ++r) l_run[r] += psum[r];

        // ---- stage P through LDS (wave-private) to A layout ----
        #pragma unroll
        for (int st = 0; st < 2; ++st)
            #pragma unroll
            for (int r = 0; r < 8; ++r)
                Plds[wave * 512 + (r + 8 * half) * 32 + st * 16 + lr] =
                    f2bfbits(sacc[st][r]);
        Frag16 pa;
        {
            const u16* prow = &Plds[wave * 512 + lr * 32];
            pa.h[0] = *(const ushort8*)(prow + half * 8);
            pa.h[1] = *(const ushort8*)(prow + 16 + half * 8);
        }

        // ---- O += P V  (V from transposed LDS: contiguous B-fragments) ----
        #pragma unroll
        for (int g = 0; g < 4; ++g) {
            const u16* vrow = &Vt[buf][(g * 16 + lr) * 32];
            Frag16 bv;
            bv.h[0] = *(const ushort8*)(vrow + half * 8);
            bv.h[1] = *(const ushort8*)(vrow + 16 + half * 8);
            oacc[g] = __builtin_amdgcn_wmma_f32_16x16x32_bf16(
                false, pa.v, false, bv.v, (short)0, oacc[g], false, false);
        }
        WAIT_ASYNC();
        __syncthreads();
    }

    // ---- finalize: O /= rowsum, write ctx bf16 ----
    #pragma unroll
    for (int g = 0; g < 4; ++g)
        #pragma unroll
        for (int r = 0; r < 8; ++r) {
            int qrow = qrowbase + r + 8 * half;
            float o = oacc[g][r] / l_run[r];
            ctx[((size_t)(b * Q_ + qrow)) * HID + h * 64 + g * 16 + lr] = f2bfbits(o);
        }
}

// ---------------------------------------------------------------------------
// present_key/value = concat(past,new)[:, :, Q:, :] == new rows 64..255 (fp32)
// ---------------------------------------------------------------------------
__global__ void copy_present_k(const float* __restrict__ kn,
                               const float* __restrict__ vn,
                               float* __restrict__ pk,
                               float* __restrict__ pv)
{
    int idx = blockIdx.x * 256 + threadIdx.x;
    int total = B_ * H_ * C_ * D_;
    if (idx >= total) return;
    int d = idx & 63;
    int r = (idx >> 6) % C_;
    int h = (idx / (64 * C_)) % H_;
    int b = idx / (64 * C_ * H_);
    size_t src = ((size_t)(b * Q_ + 64 + r)) * HID + h * 64 + d;
    pk[idx] = kn[src];
    pv[idx] = vn[src];
}

// ---------------------------------------------------------------------------
extern "C" void kernel_launch(void* const* d_in, const int* in_sizes, int n_in,
                              void* d_out, int out_size, void* d_ws, size_t ws_size,
                              hipStream_t stream)
{
    const float* hidden  = (const float*)d_in[0];
    const float* mask    = (const float*)d_in[1];
    const float* past_k  = (const float*)d_in[2];
    const float* past_v  = (const float*)d_in[3];
    const float* cross_k = (const float*)d_in[4];
    const float* cross_v = (const float*)d_in[5];
    const float* q_w  = (const float*)d_in[6];  const float* q_b  = (const float*)d_in[7];
    const float* k_w  = (const float*)d_in[8];  const float* k_b  = (const float*)d_in[9];
    const float* v_w  = (const float*)d_in[10]; const float* v_b  = (const float*)d_in[11];
    const float* o_w  = (const float*)d_in[12]; const float* o_b  = (const float*)d_in[13];
    const float* cq_w = (const float*)d_in[14]; const float* cq_b = (const float*)d_in[15];
    const float* co_w = (const float*)d_in[16]; const float* co_b = (const float*)d_in[17];
    const float* up_w = (const float*)d_in[18]; const float* up_b = (const float*)d_in[19];
    const float* dn_w = (const float*)d_in[20]; const float* dn_b = (const float*)d_in[21];
    const float* ln1w = (const float*)d_in[22]; const float* ln1b = (const float*)d_in[23];
    const float* ln2w = (const float*)d_in[24]; const float* ln2b = (const float*)d_in[25];
    const float* ln3w = (const float*)d_in[26]; const float* ln3b = (const float*)d_in[27];

    float* out_h  = (float*)d_out;
    float* out_pk = out_h + (size_t)BQ * HID;
    float* out_pv = out_pk + (size_t)B_ * H_ * C_ * D_;

    // ---- workspace carve-up ----
    char* p = (char*)d_ws;
    auto give = [&](size_t bytes) { char* r = p; p += (bytes + 255) & ~(size_t)255; return r; };
    u16*  x_bf   = (u16*)give((size_t)BQ * HID * 2);
    u16*  wq     = (u16*)give((size_t)HID * HID * 2);
    u16*  wk     = (u16*)give((size_t)HID * HID * 2);
    u16*  wv     = (u16*)give((size_t)HID * HID * 2);
    u16*  wo     = (u16*)give((size_t)HID * HID * 2);
    u16*  wcq    = (u16*)give((size_t)HID * HID * 2);
    u16*  wco    = (u16*)give((size_t)HID * HID * 2);
    u16*  wup    = (u16*)give((size_t)I_ * HID * 2);
    u16*  wdn    = (u16*)give((size_t)HID * I_ * 2);
    float* qbuf  = (float*)give((size_t)BQ * HID * 4);
    float* knew  = (float*)give((size_t)BQ * HID * 4);
    float* vnew  = (float*)give((size_t)BQ * HID * 4);
    u16*  kcache = (u16*)give((size_t)B_ * H_ * E_ * D_ * 2);
    u16*  vcache = (u16*)give((size_t)B_ * H_ * E_ * D_ * 2);
    u16*  ctx_bf = (u16*)give((size_t)BQ * HID * 2);
    float* h1    = (float*)give((size_t)BQ * HID * 4);
    float* h2    = (float*)give((size_t)BQ * HID * 4);
    u16*  up_bf  = (u16*)knew;   // alias: knew+vnew (21 MB) reused after attn1 deps

    dim3 blk(256);
    dim3 gProj(BQ / 64, HID / 128);
    dim3 gUp(BQ / 64, I_ / 128);
    dim3 gAttn(2, H_, B_);
    auto cvt = [&](const float* s, u16* d, size_t n) {
        int n8 = (int)(n / 8);
        cvt_bf16_k<<<(n8 + 255) / 256, blk, 0, stream>>>(s, d, n8);
    };

    // ---- one-shot weight conversions ----
    cvt(q_w,  wq,  (size_t)HID * HID);
    cvt(k_w,  wk,  (size_t)HID * HID);
    cvt(v_w,  wv,  (size_t)HID * HID);
    cvt(o_w,  wo,  (size_t)HID * HID);
    cvt(cq_w, wcq, (size_t)HID * HID);
    cvt(co_w, wco, (size_t)HID * HID);
    cvt(up_w, wup, (size_t)I_ * HID);
    cvt(dn_w, wdn, (size_t)HID * I_);

    // ---- self-attention block ----
    layernorm_k<<<BQ, blk, 0, stream>>>(hidden, ln1w, ln1b, x_bf, HID);
    gemm_bf16_k<<<gProj, blk, 0, stream>>>(x_bf, wq, q_b, nullptr, qbuf, nullptr,
                                           BQ, HID, HID, SCALE_, OUT_F32);
    gemm_bf16_k<<<gProj, blk, 0, stream>>>(x_bf, wk, k_b, nullptr, knew, nullptr,
                                           BQ, HID, HID, 1.0f, OUT_F32);
    gemm_bf16_k<<<gProj, blk, 0, stream>>>(x_bf, wv, v_b, nullptr, vnew, nullptr,
                                           BQ, HID, HID, 1.0f, OUT_F32);
    {
        int n = B_ * H_ * C_ * D_;
        copy_present_k<<<(n + 255) / 256, blk, 0, stream>>>(knew, vnew, out_pk, out_pv);
    }
    {
        int n = B_ * H_ * LSELF * 8;
        build_cache_self_k<<<(n + 255) / 256, blk, 0, stream>>>(past_k, past_v, knew, vnew,
                                                                kcache, vcache);
    }
    attn_k<<<gAttn, blk, 0, stream>>>(qbuf, kcache, vcache, mask, ctx_bf, LSELF);
    gemm_bf16_k<<<gProj, blk, 0, stream>>>(ctx_bf, wo, o_b, hidden, h1, nullptr,
                                           BQ, HID, HID, 1.0f, OUT_F32);

    // ---- cross-attention block ----
    layernorm_k<<<BQ, blk, 0, stream>>>(h1, ln2w, ln2b, x_bf, HID);
    gemm_bf16_k<<<gProj, blk, 0, stream>>>(x_bf, wcq, cq_b, nullptr, qbuf, nullptr,
                                           BQ, HID, HID, SCALE_, OUT_F32);
    cvt(cross_k, kcache, (size_t)B_ * H_ * E_ * D_);
    cvt(cross_v, vcache, (size_t)B_ * H_ * E_ * D_);
    attn_k<<<gAttn, blk, 0, stream>>>(qbuf, kcache, vcache, nullptr, ctx_bf, E_);
    gemm_bf16_k<<<gProj, blk, 0, stream>>>(ctx_bf, wco, co_b, h1, h2, nullptr,
                                           BQ, HID, HID, 1.0f, OUT_F32);

    // ---- MLP block (GELU fused into up-GEMM epilogue) ----
    layernorm_k<<<BQ, blk, 0, stream>>>(h2, ln3w, ln3b, x_bf, HID);
    gemm_bf16_k<<<gUp, blk, 0, stream>>>(x_bf, wup, up_b, nullptr, nullptr, up_bf,
                                         BQ, I_, HID, 1.0f, OUT_BF16 | DO_GELU);
    gemm_bf16_k<<<gProj, blk, 0, stream>>>(up_bf, wdn, dn_b, h2, out_h, nullptr,
                                           BQ, HID, I_, 1.0f, OUT_F32);

    (void)in_sizes; (void)n_in; (void)out_size; (void)ws_size;
}
